// Gemma3Attention_79328045957650
// MI455X (gfx1250) — matrix operand order repaired and checked by reading, not compiled, not run on hardware
//
#include <hip/hip_runtime.h>
#include <hip/hip_bf16.h>

// ---------------------------------------------------------------------------
// Gemma3 sliding-window attention prefill for gfx1250 (MI455X).
// All matmuls via v_wmma_f32_16x16x32_f16 (wave32).
// A/B/P fragments are stored in natural k order so every fragment is ONE contiguous
// 32B-aligned v16h load (ds_load_b128 x2 / global_load_b128 x2, no v_mov_b16).
// V and all weights are pre-transposed so contraction dims are contiguous.
// ---------------------------------------------------------------------------

typedef __attribute__((ext_vector_type(16))) _Float16 v16h;
typedef __attribute__((ext_vector_type(8)))  _Float16 v8h;
typedef __attribute__((ext_vector_type(8)))  float    v8f;

#define S_LEN   4096
#define HID     2560
#define NH      8
#define NKVH    4
#define HD      256
#define WINDOW  1024
#define QDIM    (NH * HD)    // 2048
#define KVDIM   (NKVH * HD)  // 1024
#define SCALING 0.0625f      // 256^-0.5

__device__ __forceinline__ v8f wmma_f16(v16h a, v16h b, v8f c) {
  return __builtin_amdgcn_wmma_f32_16x16x32_f16(
      /*neg_a=*/false, a, /*neg_b=*/false, b,
      /*c_mod=*/(short)0, c, /*reuse_a=*/false, /*reuse_b=*/false);
}

__device__ __forceinline__ v8f vzero8() {
  v8f z;
#pragma unroll
  for (int i = 0; i < 8; ++i) z[i] = 0.0f;
  return z;
}

__device__ __forceinline__ v16h mk16(v8h lo, v8h hi) {
  v16h a;
#pragma unroll
  for (int j = 0; j < 8; ++j) { a[j] = lo[j]; a[j + 8] = hi[j]; }
  return a;
}

// ---------------------------------------------------------------------------
// f32 -> f16 conversion (row-major copy, for the activation matrix)
// ---------------------------------------------------------------------------
__global__ void f32_to_f16_kernel(const float* __restrict__ src,
                                  _Float16* __restrict__ dst, size_t n) {
  size_t i = (size_t)blockIdx.x * blockDim.x + threadIdx.x;
  if (i < n) dst[i] = (_Float16)src[i];
}

// ---------------------------------------------------------------------------
// Transpose (+ optional f32->f16 convert): out[C][R] = in[R][C].
// 32x32 LDS tile, 256 threads. R, C multiples of 32.
// ---------------------------------------------------------------------------
template <typename InT>
__global__ __launch_bounds__(256)
void transpose_to_f16_kernel(const InT* __restrict__ in,
                             _Float16* __restrict__ out, int R, int C) {
  __shared__ _Float16 tile[32][33];
  const int c0 = blockIdx.x * 32, r0 = blockIdx.y * 32;
  const int tx = threadIdx.x & 31, ty = threadIdx.x >> 5;
#pragma unroll
  for (int i = ty; i < 32; i += 8)
    tile[i][tx] = (_Float16)in[(size_t)(r0 + i) * C + c0 + tx];
  __syncthreads();
#pragma unroll
  for (int i = ty; i < 32; i += 8)
    out[(size_t)(c0 + i) * R + r0 + tx] = tile[tx][i];
}

// ---------------------------------------------------------------------------
// Tiled WMMA GEMM: C[MxN] = A[MxK](f16, row-major) * Bt[NxK](f16, B transposed)
// Block tile 64x64, 4 waves (128 threads); wave w owns rows [w*16, w*16+16).
// Ash rows hold the 32 k-values in natural order (lane half h reads k 16h..16h+15) so an
// A fragment is one contiguous v16h; BshT rows are one N-column's 32 k-values,
// so a B fragment is one contiguous v16h. All staging is v8h vector ops.
// ---------------------------------------------------------------------------
template <typename OutT>
__global__ __launch_bounds__(128)
void gemm_wmma_kernel(const _Float16* __restrict__ A,
                      const _Float16* __restrict__ Bt,
                      OutT* __restrict__ C, int M, int N, int K) {
  constexpr int AS = 48;  // halves per A row in LDS (96B: keeps 32B alignment)
  constexpr int BT = 48;  // halves per B^T row in LDS
  __shared__ __align__(32) _Float16 Ash[64 * AS];
  __shared__ __align__(32) _Float16 BshT[64 * BT];

  const int t    = threadIdx.x;
  const int lane = t & 31;
  const int w    = t >> 5;
  const int l15  = lane & 15;
  const int kb   = (lane & 16) ? 8 : 0;    // A-fragment K sub-base
  const int kb2  = (lane & 16) ? 16 : 0;   // B-fragment K sub-base
  const int hoff = (lane & 16) ? 8 : 0;    // C row half-offset
  const int n0 = blockIdx.x * 64;
  const int m0 = blockIdx.y * 64;

  // staging coordinates
  const int arow = t >> 1;            // 64 rows, 2 threads/row
  const int akc  = (t & 1) * 16;      // which 16 k-values this thread loads
  const int bcol = t >> 1;            // 64 cols, 2 threads/col
  const int bkc  = (t & 1) * 16;

  const _Float16* ap = A  + (size_t)(m0 + arow) * K + akc;
  const _Float16* bp = Bt + (size_t)(n0 + bcol) * K + bkc;

  v8f acc[4];
#pragma unroll
  for (int i = 0; i < 4; ++i) acc[i] = vzero8();

  for (int k0 = 0; k0 < K; k0 += 32) {
    // ---- stage A tile (natural k order: lo->base, hi->base+8)
    {
      v8h lo = *(const v8h*)(ap + k0);
      v8h hi = *(const v8h*)(ap + k0 + 8);
      // akc==0 holds k0..15 : k0-7 ->pos0,  k8-15 ->pos8
      // akc==16 holds k16..31: k16-23->pos16, k24-31->pos24
      const int base = arow * AS + akc;   // 0 or 16
      *(v8h*)&Ash[base]     = lo;
      *(v8h*)&Ash[base + 8] = hi;
    }
    // ---- stage Bt tile (natural k order is already fragment order for B)
    {
      v8h lo = *(const v8h*)(bp + k0);
      v8h hi = *(const v8h*)(bp + k0 + 8);
      *(v8h*)&BshT[bcol * BT + bkc]     = lo;
      *(v8h*)&BshT[bcol * BT + bkc + 8] = hi;
    }
    if (k0 + 32 < K) {  // prefetch next K tile (global_prefetch_b8)
      __builtin_prefetch(ap + k0 + 32, 0, 0);
      __builtin_prefetch(bp + k0 + 32, 0, 0);
    }
    __syncthreads();

    // ---- fragments: single contiguous v16h loads
    v16h af = *(const v16h*)&Ash[(w * 16 + l15) * AS + kb * 2];
#pragma unroll
    for (int nt = 0; nt < 4; ++nt) {
      v16h bf = *(const v16h*)&BshT[(nt * 16 + l15) * BT + kb2];
      acc[nt] = wmma_f16(af, bf, acc[nt]);
    }
    __syncthreads();
  }

  // ---- store C (16x16 f32 acc layout: VGPR r -> row r + hoff, col = lane&15)
#pragma unroll
  for (int nt = 0; nt < 4; ++nt) {
#pragma unroll
    for (int r = 0; r < 8; ++r) {
      const int row = m0 + w * 16 + r + hoff;
      const int col = n0 + nt * 16 + l15;
      C[(size_t)row * N + col] = (OutT)acc[nt][r];
    }
  }
}

// ---------------------------------------------------------------------------
// RMSNorm (scale = 1 + w, eps 1e-6) + RoPE (HF layout: emb = cat(freqs,freqs))
// One wave per (token, head) row of 256. Lane l holds d = l*8 .. l*8+7;
// rotate_half partner d^128 lives in lane l^16 -> wave32 __shfl_xor(.,16).
// ---------------------------------------------------------------------------
__global__ __launch_bounds__(128)
void rmsnorm_rope_kernel(const _Float16* __restrict__ x,
                         const float* __restrict__ w,
                         const float* __restrict__ cosp,
                         const float* __restrict__ sinp,
                         _Float16* __restrict__ out, int nheads) {
  const int idx  = blockIdx.x * 4 + (threadIdx.x >> 5);
  const int lane = threadIdx.x & 31;
  const int s  = idx / nheads;
  const int hh = idx % nheads;
  const size_t rowbase = (size_t)s * nheads * HD + (size_t)hh * HD;
  const int dbase = lane * 8;

  v8h xv = *(const v8h*)(x + rowbase + dbase);
  float xf[8];
  float ss = 0.0f;
#pragma unroll
  for (int j = 0; j < 8; ++j) { xf[j] = (float)xv[j]; ss += xf[j] * xf[j]; }
#pragma unroll
  for (int off = 16; off > 0; off >>= 1) ss += __shfl_xor(ss, off, 32);
  const float rinv = rsqrtf(ss * (1.0f / 256.0f) + 1e-6f);

  const float sign = (lane < 16) ? -1.0f : 1.0f;  // d<128 gets -x[d+128]*sin
  v8h ov;
#pragma unroll
  for (int j = 0; j < 8; ++j) {
    const int d = dbase + j;
    const float xn = xf[j] * rinv * (1.0f + w[d]);
    const float pr = __shfl_xor(xn, 16, 32);  // normalized partner at d^128
    const float o = xn * cosp[(size_t)s * HD + d] + sign * pr * sinp[(size_t)s * HD + d];
    ov[j] = (_Float16)o;
  }
  *(v8h*)(out + rowbase + dbase) = ov;
}

// ---------------------------------------------------------------------------
// Flash-style sliding-window attention.
// grid = (S/16, H), block = 128 (4 waves). Block owns 16 queries of head h.
// Per 64-key tile: wave w computes 16x16 score slice (8 WMMAs over D=256),
// scores -> LDS, online softmax (probs written in fragment order), rescale,
// P x V (wave w owns D-slice w*64; V is pre-transposed so B fragments are
// contiguous v16h global loads). Output f16 laid out [S][H*D].
// ---------------------------------------------------------------------------
__global__ __launch_bounds__(128)
void attn_kernel(const _Float16* __restrict__ q,
                 const _Float16* __restrict__ k,
                 const _Float16* __restrict__ vT,  // [KVDIM][S_LEN]
                 _Float16* __restrict__ out) {
  constexpr int SSTR = 68;  // padded f32 score row stride
  __shared__ __align__(32) float    sS[16 * SSTR];
  __shared__ __align__(32) _Float16 sP[16 * 64];  // fragment-order probs
  __shared__ float sM[16], sL[16], sCorr[16];

  const int t    = threadIdx.x;
  const int lane = t & 31;
  const int w    = t >> 5;
  const int l15  = lane & 15;
  const int kb   = (lane & 16) ? 8 : 0;
  const int kb2  = (lane & 16) ? 16 : 0;
  const int hoff = (lane & 16) ? 8 : 0;

  const int h   = blockIdx.y;
  const int q0  = blockIdx.x * 16;
  const int kvh = h >> 1;  // GQA: rep = H/KVH = 2

  if (t < 16) { sM[t] = -1e30f; sL[t] = 0.0f; }

  // ---- load the 16x256 Q block into fragments once (8 D-chunks of 32)
  v16h qf[8];
  {
    const _Float16* qp = q + (size_t)(q0 + l15) * QDIM + (size_t)h * HD;
#pragma unroll
    for (int dc = 0; dc < 8; ++dc)
      qf[dc] = mk16(*(const v8h*)(qp + dc * 32 + kb * 2),
                    *(const v8h*)(qp + dc * 32 + kb * 2 + 8));
  }

  v8f accO[4];
#pragma unroll
  for (int i = 0; i < 4; ++i) accO[i] = vzero8();
  __syncthreads();

  int kfirst = q0 - (WINDOW - 1);
  if (kfirst < 0) kfirst = 0;

  for (int kk = (kfirst >> 6) << 6; kk <= q0 + 15; kk += 64) {
    // ---- scores: wave w -> keys kk + w*16 .. +15  (S = Q * K^T)
    {
      const int key = kk + w * 16 + l15;
      const _Float16* kp = k + (size_t)key * KVDIM + (size_t)kvh * HD;
      v8f sacc = vzero8();
#pragma unroll
      for (int dc = 0; dc < 8; ++dc) {
        // B fragment 32(d) x 16(key): contiguous 16 halves per lane
        v16h bf = *(const v16h*)(kp + dc * 32 + kb2);
        sacc = wmma_f16(qf[dc], bf, sacc);
      }
#pragma unroll
      for (int r = 0; r < 8; ++r) {
        const int row = r + hoff;
        const int qi = q0 + row;
        const int ki = kk + w * 16 + l15;
        const bool ok = (qi >= ki) && (ki >= qi - (WINDOW - 1));
        sS[row * SSTR + w * 16 + l15] = ok ? sacc[r] * SCALING : -1e30f;
      }
    }
    __syncthreads();

    // ---- online softmax, one thread per query row; probs written in
    // natural key order (contiguous A-fragment order).
    if (t < 16) {
      const int row = t;
      const float m_old = sM[row];
      float m = m_old;
      for (int j = 0; j < 64; ++j) m = fmaxf(m, sS[row * SSTR + j]);
      float corr, lsum = 0.0f;
      if (m < -1e29f) {  // row entirely masked so far
        corr = 1.0f;
        for (int j = 0; j < 64; ++j) sP[row * 64 + j] = (_Float16)0.0f;
      } else {
        corr = __expf(m_old - m);
        for (int j = 0; j < 64; ++j) {
          const float p = __expf(sS[row * SSTR + j] - m);
          lsum += p;
          const int pos = j;
          sP[row * 64 + pos] = (_Float16)p;
        }
        sM[row] = m;
      }
      sL[row] = sL[row] * corr + lsum;
      sCorr[row] = corr;
    }
    __syncthreads();

    // ---- rescale running output, then accO += P * V (wave D-slice w*64)
    float c8[8];
#pragma unroll
    for (int r = 0; r < 8; ++r) c8[r] = sCorr[r + hoff];
#pragma unroll
    for (int nt = 0; nt < 4; ++nt)
#pragma unroll
      for (int r = 0; r < 8; ++r) accO[nt][r] *= c8[r];

#pragma unroll
    for (int kc = 0; kc < 2; ++kc) {
      v16h pf = *(const v16h*)&sP[l15 * 64 + kc * 32 + kb * 2];
#pragma unroll
      for (int nt = 0; nt < 4; ++nt) {
        const int dcol = w * 64 + nt * 16 + l15;
        // V^T: contraction (keys) contiguous -> one v16h global load
        v16h vb = *(const v16h*)(vT + (size_t)(kvh * HD + dcol) * S_LEN +
                                 kk + kc * 32 + kb2);
        accO[nt] = wmma_f16(pf, vb, accO[nt]);
      }
    }
  }

  // ---- finalize: divide by l, store f16 into [S][H*D]
  float il[8];
#pragma unroll
  for (int r = 0; r < 8; ++r) il[r] = 1.0f / sL[r + hoff];
#pragma unroll
  for (int nt = 0; nt < 4; ++nt) {
#pragma unroll
    for (int r = 0; r < 8; ++r) {
      const int row = q0 + r + hoff;
      const int col = h * HD + w * 64 + nt * 16 + l15;
      out[(size_t)row * QDIM + col] = (_Float16)(accO[nt][r] * il[r]);
    }
  }
}

// ---------------------------------------------------------------------------
// Launch
// ---------------------------------------------------------------------------
extern "C" void kernel_launch(void* const* d_in, const int* in_sizes, int n_in,
                              void* d_out, int out_size, void* d_ws, size_t ws_size,
                              hipStream_t stream) {
  (void)in_sizes; (void)n_in; (void)out_size; (void)ws_size;

  const float* hs   = (const float*)d_in[0];  // (1, S, HID)
  const float* cosp = (const float*)d_in[1];  // (1,1,S,D)
  const float* sinp = (const float*)d_in[2];
  const float* wq   = (const float*)d_in[3];  // (HID, 2048)
  const float* wk   = (const float*)d_in[4];  // (HID, 1024)
  const float* wv   = (const float*)d_in[5];
  const float* wo   = (const float*)d_in[6];  // (2048, HID)
  const float* qnw  = (const float*)d_in[7];  // (D)
  const float* knw  = (const float*)d_in[8];
  float* out = (float*)d_out;                 // (1, S, HID) f32

  char* ws = (char*)d_ws;
  size_t off = 0;
  auto carve = [&](size_t bytes) { char* p = ws + off; off += (bytes + 255) & ~(size_t)255; return p; };
  _Float16* hs16  = (_Float16*)carve((size_t)S_LEN * HID  * 2);  // [S][HID]
  _Float16* wqT   = (_Float16*)carve((size_t)HID  * QDIM  * 2);  // [2048][HID]
  _Float16* wkT   = (_Float16*)carve((size_t)HID  * KVDIM * 2);  // [1024][HID]
  _Float16* wvT   = (_Float16*)carve((size_t)HID  * KVDIM * 2);  // [1024][HID]
  _Float16* woT   = (_Float16*)carve((size_t)QDIM * HID   * 2);  // [HID][2048]
  _Float16* qbuf  = (_Float16*)carve((size_t)S_LEN * QDIM  * 2); // [S][2048]
  _Float16* kbuf  = (_Float16*)carve((size_t)S_LEN * KVDIM * 2); // [S][1024]
  _Float16* vbuf  = (_Float16*)carve((size_t)S_LEN * KVDIM * 2); // [S][1024]
  _Float16* qrbuf = (_Float16*)carve((size_t)S_LEN * QDIM  * 2);
  _Float16* krbuf = (_Float16*)carve((size_t)S_LEN * KVDIM * 2);
  _Float16* vTbuf = (_Float16*)carve((size_t)S_LEN * KVDIM * 2); // [1024][S]
  _Float16* att16 = (_Float16*)carve((size_t)S_LEN * QDIM  * 2); // [S][2048]

  // convert activation; convert+transpose all weights
  {
    size_t n = (size_t)S_LEN * HID;
    f32_to_f16_kernel<<<(unsigned)((n + 255) / 256), 256, 0, stream>>>(hs, hs16, n);
  }
  transpose_to_f16_kernel<float><<<dim3(QDIM / 32,  HID / 32),  256, 0, stream>>>(wq, wqT, HID, QDIM);
  transpose_to_f16_kernel<float><<<dim3(KVDIM / 32, HID / 32),  256, 0, stream>>>(wk, wkT, HID, KVDIM);
  transpose_to_f16_kernel<float><<<dim3(KVDIM / 32, HID / 32),  256, 0, stream>>>(wv, wvT, HID, KVDIM);
  transpose_to_f16_kernel<float><<<dim3(HID / 32,  QDIM / 32),  256, 0, stream>>>(wo, woT, QDIM, HID);

  // QKV projections (A row-major, B pre-transposed)
  gemm_wmma_kernel<_Float16><<<dim3(QDIM / 64,  S_LEN / 64), 128, 0, stream>>>(
      hs16, wqT, qbuf, S_LEN, QDIM, HID);
  gemm_wmma_kernel<_Float16><<<dim3(KVDIM / 64, S_LEN / 64), 128, 0, stream>>>(
      hs16, wkT, kbuf, S_LEN, KVDIM, HID);
  gemm_wmma_kernel<_Float16><<<dim3(KVDIM / 64, S_LEN / 64), 128, 0, stream>>>(
      hs16, wvT, vbuf, S_LEN, KVDIM, HID);

  // per-head RMSNorm + RoPE for Q and K; transpose V for the PV stage
  rmsnorm_rope_kernel<<<(S_LEN * NH)   / 4, 128, 0, stream>>>(qbuf, qnw, cosp, sinp, qrbuf, NH);
  rmsnorm_rope_kernel<<<(S_LEN * NKVH) / 4, 128, 0, stream>>>(kbuf, knw, cosp, sinp, krbuf, NKVH);
  transpose_to_f16_kernel<_Float16><<<dim3(KVDIM / 32, S_LEN / 32), 256, 0, stream>>>(
      vbuf, vTbuf, S_LEN, KVDIM);

  // sliding-window flash attention
  attn_kernel<<<dim3(S_LEN / 16, NH), 128, 0, stream>>>(qrbuf, krbuf, vTbuf, att16);

  // output projection -> f32 result
  gemm_wmma_kernel<float><<<dim3(HID / 64, S_LEN / 64), 128, 0, stream>>>(
      att16, woT, out, S_LEN, HID, QDIM);
}
